// CustomBidirectionalLSTM_33543694582182
// MI455X (gfx1250) — matrix operand order repaired and checked by reading, not compiled, not run on hardware
//
#include <hip/hip_runtime.h>

// Problem sizes (fixed by the reference)
#define B_  32
#define S_  1024
#define D_  256
#define H_  256
#define G4_ 1024   // 4*H
#define ROWS_ 32768 // B*S

typedef __attribute__((ext_vector_type(16))) __bf16 bf16x16;
typedef __attribute__((ext_vector_type(8)))  float  f32x8;

struct alignas(16) Chunk16 { unsigned int u[4]; };
union FragU { Chunk16 c[2]; bf16x16 v; };

__device__ __forceinline__ bf16x16 ld_frag(const __bf16* p0, const __bf16* p1) {
  FragU f;
  f.c[0] = *reinterpret_cast<const Chunk16*>(p0);
  f.c[1] = *reinterpret_cast<const Chunk16*>(p1);
  return f.v;
}

__device__ __forceinline__ f32x8 zero8() {
  f32x8 z = {0.f,0.f,0.f,0.f,0.f,0.f,0.f,0.f};
  return z;
}

__device__ __forceinline__ float sigf(float x) { return 1.f / (1.f + __expf(-x)); }

// ---------------------------------------------------------------------------
// Kernel 1: pre-gate GEMM.  P[dir][r][g] (bf16), r = b*S + s (contiguous with
// x rows), g in [0,1024).  C = x @ W_dir + b_dir, bf16 WMMA with f32 accum.
// Workgroup tile 128x128, K-chunks of 32, 8 waves as 4(M) x 2(N).
// ---------------------------------------------------------------------------
__global__ __launch_bounds__(256)
void pregate_gemm(const float* __restrict__ x,
                  const float* __restrict__ Wf, const float* __restrict__ bfw,
                  const float* __restrict__ Wr, const float* __restrict__ brw,
                  __bf16* __restrict__ P) {
  constexpr int SA = 40;   // bf16 row stride (32 data + 8 pad) -> 80B, 16B aligned
  constexpr int SB = 40;
  __shared__ __align__(16) __bf16 ldsA[128 * SA];  // [m][k]
  __shared__ __align__(16) __bf16 ldsB[128 * SB];  // [n][k] (W transposed)

  const int tid  = threadIdx.x;
  const int lane = tid & 31;
  const int wid  = tid >> 5;
  const int hi   = lane >> 4;
  const int ln   = lane & 15;
  const int waveM = wid & 3;    // 0..3 -> 32-row strip
  const int waveN = wid >> 2;   // 0..1 -> 64-col strip

  const int mTile = blockIdx.x;        // 0..255
  const int nTile = blockIdx.y;        // 0..15  (8 per direction)
  const int dir   = nTile >> 3;
  const int nBaseDir = (nTile & 7) * 128;
  const float* W    = dir ? Wr  : Wf;
  const float* bias = dir ? brw : bfw;
  const int mBase = mTile * 128;

  f32x8 acc[2][4];
#pragma unroll
  for (int a = 0; a < 2; ++a)
#pragma unroll
    for (int b = 0; b < 4; ++b) acc[a][b] = zero8();

  for (int kt = 0; kt < 8; ++kt) {
    const int kBase = kt * 32;
    // A tile: 128 rows x 32 k of x, fp32 -> bf16
#pragma unroll
    for (int i = 0; i < 4; ++i) {
      int idx = tid + i * 256;           // 0..1023 float4s
      int row = idx >> 3;
      int c4  = (idx & 7) * 4;
      const float4 v = *reinterpret_cast<const float4*>(
          x + (size_t)(mBase + row) * D_ + kBase + c4);
      __bf16* dst = ldsA + row * SA + c4;
      dst[0] = (__bf16)v.x; dst[1] = (__bf16)v.y;
      dst[2] = (__bf16)v.z; dst[3] = (__bf16)v.w;
    }
    // B tile: W[kBase..+32][nBaseDir..+128] transposed into [n][k]
#pragma unroll
    for (int i = 0; i < 16; ++i) {
      int idx = tid + i * 256;           // 0..4095
      int k = idx >> 7;
      int n = idx & 127;
      ldsB[n * SB + k] = (__bf16)W[(size_t)(kBase + k) * G4_ + nBaseDir + n];
    }
    __syncthreads();

    bf16x16 afrag[2], bfrag[4];
#pragma unroll
    for (int mt = 0; mt < 2; ++mt) {
      const __bf16* p0 = ldsA + (waveM * 32 + mt * 16 + ln) * SA + hi * 8;
      afrag[mt] = ld_frag(p0, p0 + 16);
    }
#pragma unroll
    for (int nt = 0; nt < 4; ++nt) {
      const __bf16* p0 = ldsB + (waveN * 64 + nt * 16 + ln) * SB + hi * 8;
      bfrag[nt] = ld_frag(p0, p0 + 16);
    }
#pragma unroll
    for (int mt = 0; mt < 2; ++mt)
#pragma unroll
      for (int nt = 0; nt < 4; ++nt)
        acc[mt][nt] = __builtin_amdgcn_wmma_f32_16x16x32_bf16(
            false, afrag[mt], false, bfrag[nt], (short)0, acc[mt][nt],
            false, false);
    __syncthreads();
  }

  // Epilogue: + bias, store bf16
#pragma unroll
  for (int nt = 0; nt < 4; ++nt) {
    const int g  = nBaseDir + waveN * 64 + nt * 16 + ln;
    const float bv = bias[g];
#pragma unroll
    for (int mt = 0; mt < 2; ++mt) {
      const int m0 = mBase + waveM * 32 + mt * 16 + hi * 8;
      f32x8 a = acc[mt][nt];
#pragma unroll
      for (int r = 0; r < 8; ++r) {
        P[((size_t)dir * ROWS_ + (size_t)(m0 + r)) * G4_ + g] = (__bf16)(a[r] + bv);
      }
    }
  }
}

// ---------------------------------------------------------------------------
// Kernel 2: persistent recurrence.  8 WGs = dir(2) x batch-half(2) x col-half(2).
// Each WG keeps its 256KB bf16 slice of U resident in LDS (CDNA5: 320KB/WGP)
// and scans 1024 steps.  Column pairs exchange 128-col h slices per step via
// double-buffered L2 staging + agent-scope acquire/release flags.
// ---------------------------------------------------------------------------
__global__ __launch_bounds__(256)
void lstm_recurrence(const __bf16* __restrict__ P,
                     const float* __restrict__ Uf, const float* __restrict__ Ur,
                     float* __restrict__ out,
                     __bf16* __restrict__ gbuf,          // [2][2][2 buf][2 ch][16][128]
                     unsigned int* __restrict__ flags) { // [8]
  extern __shared__ __align__(16) __bf16 smem[];
  constexpr int SU = 264;                 // k-stride (256 data + 8 pad) -> 528B
  __bf16* ldsU  = smem;                   // [512 n][SU]  own U slice (transposed)
  __bf16* hfull = smem + 512 * SU;        // [16 m][SU]   full h (bf16, A layout)

  const int tid  = threadIdx.x;
  const int lane = tid & 31;
  const int w    = tid >> 5;   // wave 0..7 -> 16-wide h subtile
  const int hi   = lane >> 4;
  const int ln   = lane & 15;

  const int wg  = blockIdx.x;             // 0..7
  const int dir = wg >> 2;
  const int bh  = (wg >> 1) & 1;
  const int ch  = wg & 1;
  const int partner = wg ^ 1;
  const int bBase = bh * 16;
  const int ownColBase = ch * 128;        // h-column range owned
  const float* U = dir ? Ur : Uf;

  // --- one-time: stage own U slice into LDS as bf16 [n][k] ---
#pragma unroll 1
  for (int i = 0; i < 512; ++i) {
    int idx = tid + i * 256;              // 131072 elements
    int q = idx >> 15;                    // gate 0..3
    int rem = idx & 32767;
    int k = rem >> 7;                     // 0..255
    int j = rem & 127;                    // coalesced along columns
    int g = q * H_ + ownColBase + j;
    ldsU[(q * 128 + j) * SU + k] = (__bf16)U[(size_t)k * G4_ + g];
  }
  for (int i = tid; i < 16 * SU; i += 256) hfull[i] = (__bf16)0.f;
  __syncthreads();

  f32x8 c = zero8();                      // cell state, resident in VGPRs

  for (int t = 0; t < S_; ++t) {
    const int s = dir ? (S_ - 1 - t) : t;

    // accumulators start at pre-gates (already include bias)
    f32x8 acc[4];
#pragma unroll
    for (int q = 0; q < 4; ++q) {
      const int g = q * H_ + ownColBase + w * 16 + ln;
      f32x8 a;
#pragma unroll
      for (int r = 0; r < 8; ++r) {
        const int b = bBase + r + hi * 8;
        a[r] = (float)P[((size_t)dir * ROWS_ + (size_t)b * S_ + s) * G4_ + g];
      }
      acc[q] = a;
    }

    // gates += h @ U   (A = hfull in LDS, B = ldsU)
#pragma unroll
    for (int kt = 0; kt < 8; ++kt) {
      const __bf16* pa = hfull + ln * SU + kt * 32 + hi * 8;
      bf16x16 af = ld_frag(pa, pa + 16);
#pragma unroll
      for (int q = 0; q < 4; ++q) {
        const __bf16* pb = ldsU + (q * 128 + w * 16 + ln) * SU + kt * 32 + hi * 8;
        bf16x16 bfv = ld_frag(pb, pb + 16);
        acc[q] = __builtin_amdgcn_wmma_f32_16x16x32_bf16(
            false, af, false, bfv, (short)0, acc[q], false, false);
      }
    }

    // activations (lane-aligned across the 4 gate accumulators)
    f32x8 hv;
#pragma unroll
    for (int r = 0; r < 8; ++r) {
      float iv = sigf(acc[0][r]);
      float fv = sigf(acc[1][r]);
      float gv = tanhf(acc[2][r]);
      float ov = sigf(acc[3][r]);
      float cn = fv * c[r] + iv * gv;
      c[r] = cn;
      hv[r] = ov * tanhf(cn);
    }

    __syncthreads();   // all waves done reading hfull for this step

    const int colH = ownColBase + w * 16 + ln;
    const int buf  = (t + 1) & 1;
    __bf16* gslot = gbuf + (size_t)((((dir * 2 + bh) * 2 + buf) * 2 + ch)) * (16 * 128);
#pragma unroll
    for (int r = 0; r < 8; ++r) {
      const int m = r + hi * 8;
      const int b = bBase + m;
      out[((size_t)b * S_ + s) * (2 * H_) + dir * H_ + colH] = hv[r];
      __bf16 hb = (__bf16)hv[r];
      hfull[m * SU + colH] = hb;          // own columns for next step
      gslot[m * 128 + (w * 16 + ln)] = hb; // staged for partner
    }

    if (t + 1 < S_) {
      // prefetch pre-gates two steps ahead
      if (t + 2 < S_) {
        const int s2 = dir ? (S_ - 1 - (t + 2)) : (t + 2);
        __builtin_prefetch(
            &P[((size_t)dir * ROWS_ + (size_t)(bBase + hi * 8) * S_ + s2) * G4_ +
               ownColBase + w * 16 + ln], 0, 1);
      }
      __threadfence();
      __syncthreads();
      if (tid == 0) {
        __hip_atomic_fetch_add(&flags[wg], 1u, __ATOMIC_RELEASE,
                               __HIP_MEMORY_SCOPE_AGENT);
        while (__hip_atomic_load(&flags[partner], __ATOMIC_ACQUIRE,
                                 __HIP_MEMORY_SCOPE_AGENT) < (unsigned)(t + 1)) {
          __builtin_amdgcn_s_sleep(2);
        }
      }
      __syncthreads();
      // pull partner's 128 h-columns into hfull
      const __bf16* pslot =
          gbuf + (size_t)((((dir * 2 + bh) * 2 + buf) * 2 + (1 - ch))) * (16 * 128);
      const int pBase = (1 - ch) * 128;
#pragma unroll
      for (int i = tid; i < 1024; i += 256) {
        int m  = i >> 6;
        int jj = (i & 63) * 2;
        unsigned v = *reinterpret_cast<const unsigned*>(pslot + m * 128 + jj);
        *reinterpret_cast<unsigned*>(hfull + m * SU + pBase + jj) = v;
      }
      __syncthreads();
    }
  }
}

// ---------------------------------------------------------------------------
extern "C" void kernel_launch(void* const* d_in, const int* in_sizes, int n_in,
                              void* d_out, int out_size, void* d_ws, size_t ws_size,
                              hipStream_t stream) {
  (void)in_sizes; (void)n_in; (void)out_size; (void)ws_size;
  const float* x   = (const float*)d_in[0];
  const float* Wf  = (const float*)d_in[1];
  const float* Uf  = (const float*)d_in[2];
  const float* bfw = (const float*)d_in[3];
  const float* Wr  = (const float*)d_in[4];
  const float* Ur  = (const float*)d_in[5];
  const float* brw = (const float*)d_in[6];
  float* out = (float*)d_out;

  // Workspace layout: P (bf16 pre-gates, 128MB) | gbuf (64KB) | flags
  __bf16* P = (__bf16*)d_ws;
  const size_t pBytes = (size_t)2 * ROWS_ * G4_ * sizeof(__bf16);
  __bf16* gbuf = (__bf16*)((char*)d_ws + pBytes);
  const size_t gBytes = (size_t)2 * 2 * 2 * 2 * 16 * 128 * sizeof(__bf16);
  unsigned int* flags = (unsigned int*)((char*)d_ws + pBytes + gBytes);

  hipMemsetAsync(flags, 0, 8 * sizeof(unsigned int), stream);

  pregate_gemm<<<dim3(256, 16), dim3(256), 0, stream>>>(x, Wf, bfw, Wr, brw, P);

  const int smemBytes = (512 + 16) * 264 * (int)sizeof(__bf16); // ~272KB < 320KB WGP LDS
  hipFuncSetAttribute((const void*)lstm_recurrence,
                      hipFuncAttributeMaxDynamicSharedMemorySize, smemBytes);
  lstm_recurrence<<<dim3(8), dim3(256), smemBytes, stream>>>(P, Uf, Ur, out,
                                                             gbuf, flags);
}
